// Graph_Attention_Union_ar_76879914598920
// MI455X (gfx1250) — matrix operand order repaired and checked by reading, not compiled, not run on hardware
//
#include <hip/hip_runtime.h>
#include <hip/hip_bf16.h>

typedef __attribute__((ext_vector_type(16))) _Float16 v16h;
typedef __attribute__((ext_vector_type(8)))  _Float16 v8h;
typedef __attribute__((ext_vector_type(8)))  float    v8f;

#define WMMA_F32_F16(a, b, c) \
  __builtin_amdgcn_wmma_f32_16x16x32_f16(false, (a), false, (b), (short)0, (c), false, false)

// Load a contiguous, 16B-aligned 16-half fragment from LDS as two b128 loads.
__device__ __forceinline__ v16h ld_frag(const _Float16* p) {
  union { v16h v; v8h h[2]; } u;
  u.h[0] = *(const v8h*)p;
  u.h[1] = *(const v8h*)(p + 8);
  return u.v;
}

// ---------------------------------------------------------------------------
// Generic WMMA GEMM: D[M,N] = epilogue( sum_k A(m,k)*B(k,n) )
//   A(m,k) = TA ? A[k*lda+m] : A[m*lda+k]
//   B(k,n) = TB ? B[n*ldb+k] : B[k*ldb+n]
// Block tile 64x128, BK=32, 256 threads = 8 wave32 waves, wave tile 32x32
// (2x2 v_wmma_f32_16x16x32_f16). Double-buffered LDS, register prefetch.
// All edge guards are BRANCHLESS: clamp index -> unconditional load -> select.
// LDS layouts are fragment-major:
//   As row m: [K0..K31]                          -> lane reads 16 contig halves
//   Bt row n: [K0..K31]                         -> lane reads 16 contig halves
// EPI: 0=none  1=+bias  2=+bias,BN,PReLU  3=BN,ReLU(no bias)  4=+bias,BN
// ---------------------------------------------------------------------------
template<int EPI, bool TA, bool TB>
__global__ void __launch_bounds__(256)
wmma_gemm(const float* __restrict__ A, const float* __restrict__ Bm, float* __restrict__ D,
          int M, int N, int K, int lda, int ldb, int ldd,
          long long sA, long long sB, long long sD,
          const float* __restrict__ bias, const float* __restrict__ gam,
          const float* __restrict__ bet, const float* __restrict__ mean,
          const float* __restrict__ var, const float* __restrict__ alpha, float eps)
{
  __shared__ _Float16 As[2][64][40];    // 80B rows: 16B-aligned fragments
  __shared__ _Float16 Bt[2][128][40];
  const int tid = threadIdx.x;
  const int bM = blockIdx.y * 64, bN = blockIdx.x * 128;
  A  += (long long)blockIdx.z * sA;
  Bm += (long long)blockIdx.z * sB;
  D  += (long long)blockIdx.z * sD;

  const int lane = tid & 31, w = tid >> 5;
  const int wm = w >> 2, wn = w & 3;          // wave tile: rows wm*32+, cols wn*32+
  const int half = lane >> 4, l16 = lane & 15;

  // staging assignments
  const int ami = tid >> 2, ci = tid & 3, aki = ci * 8;   // A: 8 consecutive k
  const int aoff = aki;                                   // row offset = k offset
  const int bki = tid >> 3, bni = (tid & 7) * 16;         // B: 16 consecutive n
  const bool amOK = (bM + ami) < M;
  const int  amIdx = amOK ? (bM + ami) : 0;               // clamped row (valid addr)

  float ra[8], rb[16];
  auto loadA = [&](int k0) {
    if (TA) {
#pragma unroll
      for (int j = 0; j < 8; ++j) {
        int k = k0 + aki + j;
        int kc = (k < K) ? k : (K - 1);
        float v = A[(long long)kc * lda + amIdx];          // unconditional, in-bounds
        ra[j] = (amOK && k < K) ? v : 0.f;
      }
    } else {
      const float* p = A + (long long)amIdx * lda + (k0 + aki);
      if (k0 + 32 <= K) {                                  // uniform full-chunk fast path
#pragma unroll
        for (int j = 0; j < 8; ++j) { float v = p[j]; ra[j] = amOK ? v : 0.f; }
      } else {
#pragma unroll
        for (int j = 0; j < 8; ++j) {
          int k = k0 + aki + j;
          int kc = (k < K) ? k : (K - 1);
          float v = A[(long long)amIdx * lda + kc];
          ra[j] = (amOK && k < K) ? v : 0.f;
        }
      }
    }
  };
  auto loadB = [&](int k0) {
    const int k = k0 + bki;
    const bool kOK = k < K;
    const int kc = kOK ? k : (K - 1);
    if (TB) {
#pragma unroll
      for (int j = 0; j < 16; ++j) {
        int n = bN + bni + j;
        int nc = (n < N) ? n : (N - 1);
        float v = Bm[(long long)nc * ldb + kc];            // unconditional
        rb[j] = (kOK && n < N) ? v : 0.f;
      }
    } else {
      const float* p = Bm + (long long)kc * ldb;
#pragma unroll
      for (int j = 0; j < 16; ++j) {
        int n = bN + bni + j;
        int nc = (n < N) ? n : (N - 1);
        float v = p[nc];                                   // unconditional
        rb[j] = (kOK && n < N) ? v : 0.f;
      }
    }
  };
  auto stage = [&](int buf) {
    v8h av;
#pragma unroll
    for (int j = 0; j < 8; ++j) av[j] = (_Float16)ra[j];
    *(v8h*)&As[buf][ami][aoff] = av;            // one 16B LDS store
#pragma unroll
    for (int j = 0; j < 16; ++j)                // transposed scalar stores
      Bt[buf][bni + j][bki] = (_Float16)rb[j];
  };

  v8f acc00 = {}, acc01 = {}, acc10 = {}, acc11 = {};
  const int nIter = (K + 31) >> 5;

  loadA(0); loadB(0); stage(0);
  __syncthreads();

  for (int it = 0; it < nIter; ++it) {
    const int cur = it & 1;
    if (it + 1 < nIter) { loadA((it + 1) * 32); loadB((it + 1) * 32); }

    v16h af0 = ld_frag(&As[cur][wm * 32 + l16][16 * half]);
    v16h af1 = ld_frag(&As[cur][wm * 32 + 16 + l16][16 * half]);
    v16h bf0 = ld_frag(&Bt[cur][wn * 32 + l16][16 * half]);
    v16h bf1 = ld_frag(&Bt[cur][wn * 32 + 16 + l16][16 * half]);

    acc00 = WMMA_F32_F16(af0, bf0, acc00);
    acc01 = WMMA_F32_F16(af0, bf1, acc01);
    acc10 = WMMA_F32_F16(af1, bf0, acc10);
    acc11 = WMMA_F32_F16(af1, bf1, acc11);

    if (it + 1 < nIter) stage(1 - cur);
    __syncthreads();
  }

  float a0 = 0.f;
  if (EPI == 2) a0 = alpha[0];
#pragma unroll
  for (int tm = 0; tm < 2; ++tm) {
#pragma unroll
    for (int tn = 0; tn < 2; ++tn) {
      const v8f* accp = (tm == 0) ? (tn == 0 ? &acc00 : &acc01)
                                  : (tn == 0 ? &acc10 : &acc11);
      int n_g = bN + wn * 32 + tn * 16 + l16;
#pragma unroll
      for (int r = 0; r < 8; ++r) {
        int m_g = bM + wm * 32 + tm * 16 + r + 8 * half;
        if (m_g < M && n_g < N) {
          float v = (*accp)[r];
          if (EPI == 1 || EPI == 2 || EPI == 4) v += bias[m_g];
          if (EPI >= 2) {
            float inv = rsqrtf(var[m_g] + eps);
            v = (v - mean[m_g]) * (gam[m_g] * inv) + bet[m_g];
          }
          if (EPI == 2) v = (v >= 0.f) ? v : a0 * v;
          if (EPI == 3) v = fmaxf(v, 0.f);
          D[(long long)m_g * ldd + n_g] = v;
        }
      }
    }
  }
}

// ---------------------------------------------------------------------------
// Implicit-GEMM 3x3 SAME conv, C=256 -> O=256 via WMMA. K = 256*9 = 2304.
// OIHW weights are row-major [O][2304]; B tile im2cols on the fly with an
// incremental (y,x) walk and BRANCHLESS border handling (clamp+select), so the
// 16 pixel loads issue as one clause. Epilogue: +bias, BN, PReLU.
// ---------------------------------------------------------------------------
__global__ void __launch_bounds__(256)
conv3x3_wmma(const float* __restrict__ Wt, const float* __restrict__ X, float* __restrict__ D,
             int Hh, int Ww,
             const float* __restrict__ bias, const float* __restrict__ gam,
             const float* __restrict__ bet, const float* __restrict__ mean,
             const float* __restrict__ var, const float* __restrict__ alpha, float eps)
{
  __shared__ _Float16 As[2][64][40];
  __shared__ _Float16 Bt[2][128][40];
  const int tid = threadIdx.x;
  const int N = Hh * Ww;                 // 625 or 169
  const int K = 2304, M = 256;
  const int bM = blockIdx.y * 64, bN = blockIdx.x * 128;
  const float* Xb = X + (long long)blockIdx.z * 256 * N;
  float* Db = D + (long long)blockIdx.z * 256 * N;

  const int lane = tid & 31, w = tid >> 5;
  const int wm = w >> 2, wn = w & 3;
  const int half = lane >> 4, l16 = lane & 15;

  const int ami = tid >> 2, ci = tid & 3, aki = ci * 8;
  const int aoff = aki;
  const int bki = tid >> 3, bni = (tid & 7) * 16;

  // (y,x) of this thread's first output pixel -- independent of K
  const int n0 = bN + bni;
  const int y0 = n0 / Ww, x0 = n0 - y0 * Ww;

  float ra[8], rb[16];
  auto loadA = [&](int k0) {
    const float* p = Wt + (long long)(bM + ami) * K + (k0 + aki);
#pragma unroll
    for (int j = 0; j < 8; ++j) ra[j] = p[j];
  };
  auto loadB = [&](int k0) {
    const int k = k0 + bki;                  // < 2304 always (K % 32 == 0)
    const int c = k / 9, t9 = k - 9 * c;
    const int ky = t9 / 3 - 1, kx = t9 - 3 * (t9 / 3) - 1;
    const float* Xc = Xb + (long long)c * N;
    int y = y0, x = x0;
#pragma unroll
    for (int j = 0; j < 16; ++j) {
      const int iy = y + ky, ix = x + kx;
      // clamp to a valid pixel, load unconditionally, select 0 on the border
      int cy = iy < 0 ? 0 : (iy >= Hh ? Hh - 1 : iy);
      int cx = ix < 0 ? 0 : (ix >= Ww ? Ww - 1 : ix);
      float v = Xc[cy * Ww + cx];
      bool ok = (y < Hh) & (iy >= 0) & (iy < Hh) & (ix >= 0) & (ix < Ww);
      rb[j] = ok ? v : 0.f;
      if (++x == Ww) { x = 0; ++y; }
    }
  };
  auto stage = [&](int buf) {
    v8h av;
#pragma unroll
    for (int j = 0; j < 8; ++j) av[j] = (_Float16)ra[j];
    *(v8h*)&As[buf][ami][aoff] = av;
#pragma unroll
    for (int j = 0; j < 16; ++j)
      Bt[buf][bni + j][bki] = (_Float16)rb[j];
  };

  v8f acc00 = {}, acc01 = {}, acc10 = {}, acc11 = {};
  const int nIter = K >> 5;              // 72

  loadA(0); loadB(0); stage(0);
  __syncthreads();

  for (int it = 0; it < nIter; ++it) {
    const int cur = it & 1;
    if (it + 1 < nIter) { loadA((it + 1) * 32); loadB((it + 1) * 32); }

    v16h af0 = ld_frag(&As[cur][wm * 32 + l16][16 * half]);
    v16h af1 = ld_frag(&As[cur][wm * 32 + 16 + l16][16 * half]);
    v16h bf0 = ld_frag(&Bt[cur][wn * 32 + l16][16 * half]);
    v16h bf1 = ld_frag(&Bt[cur][wn * 32 + 16 + l16][16 * half]);

    acc00 = WMMA_F32_F16(af0, bf0, acc00);
    acc01 = WMMA_F32_F16(af0, bf1, acc01);
    acc10 = WMMA_F32_F16(af1, bf0, acc10);
    acc11 = WMMA_F32_F16(af1, bf1, acc11);

    if (it + 1 < nIter) stage(1 - cur);
    __syncthreads();
  }

  const float a0 = alpha[0];
#pragma unroll
  for (int tm = 0; tm < 2; ++tm) {
#pragma unroll
    for (int tn = 0; tn < 2; ++tn) {
      const v8f* accp = (tm == 0) ? (tn == 0 ? &acc00 : &acc01)
                                  : (tn == 0 ? &acc10 : &acc11);
      int n_g = bN + wn * 32 + tn * 16 + l16;
#pragma unroll
      for (int r = 0; r < 8; ++r) {
        int m_g = bM + wm * 32 + tm * 16 + r + 8 * half;
        if (m_g < M && n_g < N) {
          float v = (*accp)[r] + bias[m_g];
          float inv = rsqrtf(var[m_g] + eps);
          v = (v - mean[m_g]) * (gam[m_g] * inv) + bet[m_g];
          v = (v >= 0.f) ? v : a0 * v;
          Db[(long long)m_g * N + n_g] = v;
        }
      }
    }
  }
}

// ---------------------------------------------------------------------------
// Row-wise sparsemax via bisection on tau. Row cached in registers (L<=768).
// ---------------------------------------------------------------------------
__global__ void sparsemax_rows(float* __restrict__ S, int L)
{
  float* s = S + (long long)blockIdx.x * L;
  __shared__ float red[256];
  const int t = threadIdx.x;
  float vals[3];
  int cnt = 0;
  for (int i = t; i < L; i += 256) vals[cnt++] = s[i];

  float mx = -3.4e38f;
  for (int c = 0; c < cnt; ++c) mx = fmaxf(mx, vals[c]);
  red[t] = mx; __syncthreads();
  for (int o = 128; o > 0; o >>= 1) { if (t < o) red[t] = fmaxf(red[t], red[t + o]); __syncthreads(); }
  mx = red[0]; __syncthreads();

  float lo = mx - 1.f, hi = mx;
  for (int it = 0; it < 32; ++it) {
    float mid = 0.5f * (lo + hi);
    float sum = 0.f;
    for (int c = 0; c < cnt; ++c) sum += fmaxf(vals[c] - mid, 0.f);
    red[t] = sum; __syncthreads();
    for (int o = 128; o > 0; o >>= 1) { if (t < o) red[t] += red[t + o]; __syncthreads(); }
    if (red[0] >= 1.f) lo = mid; else hi = mid;
    __syncthreads();
  }
  float tau = 0.5f * (lo + hi);
  cnt = 0;
  for (int i = t; i < L; i += 256) s[i] = fmaxf(vals[cnt++] - tau, 0.f);
}

// Row-wise softmax (in-place), one block per row.
__global__ void softmax_rows(float* __restrict__ S, int L)
{
  float* s = S + (long long)blockIdx.x * L;
  __shared__ float red[256];
  const int t = threadIdx.x;
  float mx = -3.4e38f;
  for (int i = t; i < L; i += 256) mx = fmaxf(mx, s[i]);
  red[t] = mx; __syncthreads();
  for (int o = 128; o > 0; o >>= 1) { if (t < o) red[t] = fmaxf(red[t], red[t + o]); __syncthreads(); }
  mx = red[0]; __syncthreads();
  float sum = 0.f;
  for (int i = t; i < L; i += 256) { float e = expf(s[i] - mx); s[i] = e; sum += e; }
  red[t] = sum; __syncthreads();
  for (int o = 128; o > 0; o >>= 1) { if (t < o) red[t] += red[t + o]; __syncthreads(); }
  float inv = 1.f / red[0];
  for (int i = t; i < L; i += 256) s[i] *= inv;
}

// SE pooling: cat (B,512,25,25) -> cat50 (B,512,50) = [mean over H | mean over W]
__global__ void se_pool(const float* __restrict__ cat, float* __restrict__ cat50)
{
  int idx = blockIdx.x * 256 + threadIdx.x;     // (b*512+c)*50 + i
  if (idx >= 64 * 512 * 50) return;
  int i = idx % 50; int bc = idx / 50;
  const float* p = cat + (long long)bc * 625;
  float s = 0.f;
  if (i < 25) { for (int y = 0; y < 25; ++y) s += p[y * 25 + i]; }
  else        { int y = i - 25; for (int x = 0; x < 25; ++x) s += p[y * 25 + x]; }
  cat50[idx] = s * (1.f / 25.f);
}

// SE MLP per (b,ch): 4x(50->12 concat) -> 48->50 -> 50->50.
// Writes fc6 TRANSPOSED: fc6t (B,50,512) so that fc7 reads are contiguous.
__global__ void se_mlp(const float* __restrict__ cat50,
                       const float* w0, const float* b0, const float* w1, const float* b1,
                       const float* w2, const float* b2, const float* w3, const float* b3,
                       const float* w5, const float* b5, const float* w6, const float* b6,
                       float* __restrict__ fc6t)
{
  int bc = blockIdx.x, t = threadIdx.x;         // 64 threads
  int b = bc >> 9, ch = bc & 511;
  __shared__ float sin_[50], sfcs[48], s5[50];
  const float* row = cat50 + (long long)bc * 50;
  if (t < 50) sin_[t] = row[t];
  __syncthreads();
  if (t < 48) {
    int i = t / 12, j = t % 12;
    const float* W  = (i == 0 ? w0 : i == 1 ? w1 : i == 2 ? w2 : w3);
    const float* Bv = (i == 0 ? b0 : i == 1 ? b1 : i == 2 ? b2 : b3);
    float a = Bv[j];
    for (int k = 0; k < 50; ++k) a += sin_[k] * W[j * 50 + k];
    sfcs[t] = a;
  }
  __syncthreads();
  if (t < 50) {
    float a = b5[t];
    for (int k = 0; k < 48; ++k) a += sfcs[k] * w5[t * 48 + k];
    s5[t] = a;
  }
  __syncthreads();
  if (t < 50) {
    float a = b6[t];
    for (int k = 0; k < 50; ++k) a += s5[k] * w6[t * 50 + k];
    fc6t[((long long)b * 50 + t) * 512 + ch] = a;
  }
}

// fc7: for each (b,j): 512-vec (contiguous) x W7^T -> fc7 (B,256,50)
__global__ void se_fc7(const float* __restrict__ fc6t, const float* __restrict__ w7,
                       const float* __restrict__ b7, float* __restrict__ fc7)
{
  int idx = blockIdx.x * 256 + threadIdx.x;     // (b*256+o)*50 + j
  if (idx >= 64 * 256 * 50) return;
  int j = idx % 50; int bo = idx / 50; int o = bo % 256; int b = bo / 256;
  const float* f = fc6t + ((long long)b * 50 + j) * 512;
  const float* w = w7 + (long long)o * 512;
  float a = b7[o];
  for (int ch = 0; ch < 512; ++ch) a += f[ch] * w[ch];
  fc7[idx] = a;
}

// final: out = gelu_exact( BN_f( xr * (1 + sh[y]*sw[x]) ) )
__global__ void final_gelu(const float* __restrict__ xr, const float* __restrict__ fc7,
                           const float* __restrict__ g, const float* __restrict__ bet,
                           const float* __restrict__ mean, const float* __restrict__ var,
                           float* __restrict__ out)
{
  long long idx = (long long)blockIdx.x * 256 + threadIdx.x;
  if (idx >= (long long)64 * 256 * 625) return;
  int n = (int)(idx % 625); long long bc = idx / 625; int c = (int)(bc % 256);
  int y = n / 25, x = n % 25;
  float v = xr[idx];
  const float* f7 = fc7 + bc * 50;
  v = v + v * f7[y] * f7[25 + x];
  float inv = rsqrtf(var[c] + 1e-5f);
  v = (v - mean[c]) * (g[c] * inv) + bet[c];
  out[idx] = 0.5f * v * (1.f + erff(v * 0.70710678118654752f));
}

// ---------------------------------------------------------------------------
extern "C" void kernel_launch(void* const* d_in, const int* in_sizes, int n_in,
                              void* d_out, int out_size, void* d_ws, size_t ws_size,
                              hipStream_t stream)
{
  // setup_inputs() dict order: zf, xf, then params in insertion order.
  const float* zf   = (const float*)d_in[0];   // (64,256,13,13)
  const float* xf   = (const float*)d_in[1];   // (64,256,25,25)
  const float* qw   = (const float*)d_in[2];
  const float* qb   = (const float*)d_in[3];
  const float* sw   = (const float*)d_in[4];
  const float* sb   = (const float*)d_in[5];
  const float* hmw  = (const float*)d_in[6];
  const float* hmb  = (const float*)d_in[7];
  const float* hmg  = (const float*)d_in[8];
  const float* hmbe = (const float*)d_in[9];
  const float* hmme = (const float*)d_in[10];
  const float* hmva = (const float*)d_in[11];
  const float* hma  = (const float*)d_in[12];
  const float* haw  = (const float*)d_in[13];
  const float* hab  = (const float*)d_in[14];
  const float* hag  = (const float*)d_in[15];
  const float* habe = (const float*)d_in[16];
  const float* hame = (const float*)d_in[17];
  const float* hava = (const float*)d_in[18];
  const float* haa  = (const float*)d_in[19];
  const float* gg   = (const float*)d_in[20];
  const float* gbe  = (const float*)d_in[21];
  const float* gme  = (const float*)d_in[22];
  const float* gva  = (const float*)d_in[23];
  const float* fc5w = (const float*)d_in[24];
  const float* fc5b = (const float*)d_in[25];
  const float* fc6w = (const float*)d_in[26];
  const float* fc6b = (const float*)d_in[27];
  const float* fc7w = (const float*)d_in[28];
  const float* fc7b = (const float*)d_in[29];
  const float* crw  = (const float*)d_in[30];
  const float* crb  = (const float*)d_in[31];
  const float* crg  = (const float*)d_in[32];
  const float* crbe = (const float*)d_in[33];
  const float* crme = (const float*)d_in[34];
  const float* crva = (const float*)d_in[35];
  const float* fg   = (const float*)d_in[36];
  const float* fbe  = (const float*)d_in[37];
  const float* fme  = (const float*)d_in[38];
  const float* fva  = (const float*)d_in[39];
  const float* fc0w = (const float*)d_in[40];
  const float* fc0b = (const float*)d_in[41];
  const float* fc1w = (const float*)d_in[42];
  const float* fc1b = (const float*)d_in[43];
  const float* fc2w = (const float*)d_in[44];
  const float* fc2b = (const float*)d_in[45];
  const float* fc3w = (const float*)d_in[46];
  const float* fc3b = (const float*)d_in[47];

  float* ws = (float*)d_ws;
  // Workspace arena with lifetime reuse (floats). Peak ~70.7M floats (~283 MB).
  const long long off_simA = 0;           // 25,000,000 : simx -> crossS -> xr
  const long long off_B    = 25000000LL;  // 10,240,000 : a_x -> q
  const long long off_cat  = 35240000LL;  // 20,480,000 : cat = [emb ; xg]
  const long long off_D    = 55720000LL;  //  2,768,896 : e_x -> s
  const long long off_ez   = 58488896LL;  //    692,224
  const long long off_simz = 59181120LL;  //  1,827,904
  const long long off_az   = 61009024LL;  //  2,768,896
  const long long off_zg   = 63777920LL;  //  2,768,896
  const long long off_c50  = 66546816LL;  //  1,638,400
  const long long off_f6   = 68185216LL;  //  1,638,400
  const long long off_f7   = 69823616LL;  //    819,200
  (void)ws_size; (void)in_sizes; (void)n_in; (void)out_size;

  // ---- x branch of gblock ----
  // e_x = prelu(bn(h_m_w @ xf))        M=64 N=625 K=256
  wmma_gemm<2,false,false><<<dim3(5,1,64),256,0,stream>>>(
      hmw, xf, ws+off_D, 64,625,256, 256,625,625, 0, 160000, 40000,
      hmb, hmg, hmbe, hmme, hmva, hma, 1e-5f);
  // simx = e_x^T e_x                   M=625 N=625 K=64
  wmma_gemm<0,true,false><<<dim3(5,10,64),256,0,stream>>>(
      ws+off_D, ws+off_D, ws+off_simA, 625,625,64, 625,625,625, 40000,40000,390625,
      nullptr,nullptr,nullptr,nullptr,nullptr,nullptr,0.f);
  sparsemax_rows<<<64*625,256,0,stream>>>(ws+off_simA, 625);
  // a_x = prelu(bn(conv3x3(xf)))       implicit GEMM M=256 N=625 K=2304
  conv3x3_wmma<<<dim3(5,4,64),256,0,stream>>>(
      haw, xf, ws+off_B, 25,25, hab, hag, habe, hame, hava, haa, 1e-5f);
  // xg = relu(bn_g(a_x @ simx)) -> cat rows 256..511
  wmma_gemm<3,false,false><<<dim3(5,4,64),256,0,stream>>>(
      ws+off_B, ws+off_simA, ws+off_cat+256*625, 256,625,625, 625,625,625,
      160000,390625,320000, nullptr, gg, gbe, gme, gva, nullptr, 1e-5f);

  // ---- z branch of gblock ----
  wmma_gemm<2,false,false><<<dim3(2,1,64),256,0,stream>>>(
      hmw, zf, ws+off_ez, 64,169,256, 256,169,169, 0, 43264, 10816,
      hmb, hmg, hmbe, hmme, hmva, hma, 1e-5f);
  wmma_gemm<0,true,false><<<dim3(2,3,64),256,0,stream>>>(
      ws+off_ez, ws+off_ez, ws+off_simz, 169,169,64, 169,169,169, 10816,10816,28561,
      nullptr,nullptr,nullptr,nullptr,nullptr,nullptr,0.f);
  sparsemax_rows<<<64*169,256,0,stream>>>(ws+off_simz, 169);
  conv3x3_wmma<<<dim3(2,4,64),256,0,stream>>>(
      haw, zf, ws+off_az, 13,13, hab, hag, habe, hame, hava, haa, 1e-5f);
  wmma_gemm<3,false,false><<<dim3(2,4,64),256,0,stream>>>(
      ws+off_az, ws+off_simz, ws+off_zg, 256,169,169, 169,169,169,
      43264,28561,43264, nullptr, gg, gbe, gme, gva, nullptr, 1e-5f);

  // ---- cross attention ----
  // q = q_w @ xf + q_b   (a_x region is dead)
  wmma_gemm<1,false,false><<<dim3(5,4,64),256,0,stream>>>(
      qw, xf, ws+off_B, 256,625,256, 256,625,625, 0, 160000, 160000,
      qb, nullptr,nullptr,nullptr,nullptr,nullptr, 0.f);
  // s = s_w @ zf + s_b   (e_x region is dead)
  wmma_gemm<1,false,false><<<dim3(2,4,64),256,0,stream>>>(
      sw, zf, ws+off_D, 256,169,256, 256,169,169, 0, 43264, 43264,
      sb, nullptr,nullptr,nullptr,nullptr,nullptr, 0.f);
  // crossS[k,l] = sum_c q[c,k] s[c,l]  M=625 N=169 K=256  (simx region is dead)
  wmma_gemm<0,true,false><<<dim3(2,10,64),256,0,stream>>>(
      ws+off_B, ws+off_D, ws+off_simA, 625,169,256, 625,169,169, 160000,43264,105625,
      nullptr,nullptr,nullptr,nullptr,nullptr,nullptr,0.f);
  softmax_rows<<<64*625,256,0,stream>>>(ws+off_simA, 169);
  // emb[c,k] = sum_l zg[c,l] crossS[k,l]  -> cat rows 0..255  (B transposed)
  wmma_gemm<0,false,true><<<dim3(5,4,64),256,0,stream>>>(
      ws+off_zg, ws+off_simA, ws+off_cat, 256,625,169, 169,169,625,
      43264,105625,320000, nullptr,nullptr,nullptr,nullptr,nullptr,nullptr,0.f);

  // ---- SE block ----
  se_pool<<<(64*512*50+255)/256,256,0,stream>>>(ws+off_cat, ws+off_c50);
  se_mlp<<<64*512,64,0,stream>>>(ws+off_c50,
      fc0w,fc0b, fc1w,fc1b, fc2w,fc2b, fc3w,fc3b, fc5w,fc5b, fc6w,fc6b, ws+off_f6);
  se_fc7<<<(64*256*50+255)/256,256,0,stream>>>(ws+off_f6, fc7w, fc7b, ws+off_f7);
  // xr = bn_{eps=1e-3}(se_cr_w @ cat + b)   M=256 N=625 K=512  (crossS region dead)
  wmma_gemm<4,false,false><<<dim3(5,4,64),256,0,stream>>>(
      crw, ws+off_cat, ws+off_simA, 256,625,512, 512,625,625, 0, 320000, 160000,
      crb, crg, crbe, crme, crva, nullptr, 1e-3f);

  // ---- final fuse: xr*(1+sh*sw) -> BN_f -> exact GELU ----
  final_gelu<<<(64*256*625+255)/256,256,0,stream>>>(
      ws+off_simA, ws+off_f7, fg, fbe, fme, fva, (float*)d_out);
}
